// DGCNN_pseg_46514495816099
// MI455X (gfx1250) — compile-verified, matrix operand order checked by reading, not run on hardware
//
#include <hip/hip_runtime.h>
#include <hip/hip_bf16.h>
#include <cstdint>
#include <cstddef>

// ---------------------------------------------------------------------------
// DGCNN part-seg forward for MI455X (gfx1250, wave32, WMMA).
// GEMMs: v_wmma_f32_16x16x32_bf16, bf16 operands staged in LDS.
// Tile staging: global_load_async_to_lds_b128 (double-buffered, ASYNCcnt
// pipelined) when the toolchain declares the builtin; plain b128 copies else.
// ---------------------------------------------------------------------------

#define B_  16
#define P_  2048
#define K_  20
#define BP_ (B_ * P_)          // 32768
#define E_  (BP_ * K_)         // 655360
#define SLOPE_ 0.2f
#define EPS_ 1e-5f

typedef __attribute__((ext_vector_type(16))) __bf16 v16bf;
typedef __attribute__((ext_vector_type(8)))  float  v8f;
typedef int v4i __attribute__((vector_size(16)));   // matches builtin proto

#ifdef __has_builtin
#if __has_builtin(__builtin_amdgcn_global_load_async_to_lds_b128) && \
    __has_builtin(__builtin_amdgcn_s_wait_asynccnt)
#define USE_ASYNC 1
#endif
#endif
#ifndef USE_ASYNC
#define USE_ASYNC 0
#endif

// fp32 -> bf16 round-to-nearest-even
__device__ inline __bf16 f2bf(float f) {
  union { float f; uint32_t u; } x; x.f = f;
  uint32_t r = x.u + 0x7FFFu + ((x.u >> 16) & 1u);
  union { uint16_t s; __bf16 b; } y; y.s = (uint16_t)(r >> 16);
  return y.b;
}

// 16-element bf16 fragment: 8 halves at p, 8 at p+16 (CDNA5 16-bit A/B VGPR
// layout: K 0-7/16-23 for lanes 0-15, K 8-15/24-31 for lanes 16-31 via +8).
__device__ inline v16bf frag_load(const __bf16* p) {
  union { uint4 u[2]; v16bf v; } fr;
  fr.u[0] = *(const uint4*)(p);
  fr.u[1] = *(const uint4*)(p + 16);
  return fr.v;
}

// Copy a 64x32 bf16 tile (rows row0.., cols kk..kk+31 of a row-major matrix
// with leading dim ld) into LDS dst[64][40].  4KB = 256 x b128; 2 per thread.
__device__ inline void copy_tile(const __bf16* __restrict__ src, int ld,
                                 int row0, int kk, __bf16 (*dst)[40], int tid) {
#pragma unroll
  for (int q = 0; q < 2; ++q) {
    int c = tid + 128 * q;
    int r = c >> 2, sub = (c & 3) * 8;
    const __bf16* gp = src + (size_t)(row0 + r) * ld + kk + sub;
#if USE_ASYNC
    __builtin_amdgcn_global_load_async_to_lds_b128(
        (__attribute__((address_space(1))) v4i*)gp,
        (__attribute__((address_space(3))) v4i*)&dst[r][sub], 0, 0);
#else
    *(uint4*)(&dst[r][sub]) = *(const uint4*)gp;
#endif
  }
}

// Edge-feature element: virtual A row r of (E x 2Cf): [xi, xj - xi]
__device__ inline float edge_elem(int r, int c, const float* __restrict__ X,
                                  const int* __restrict__ nidx,
                                  int Cf, int ldx, int coff) {
  int bp = r / K_;                 // b*P + p
  int b  = bp >> 11;               // P_ == 2048
  int nb = nidx[r];
  const float* xi = X + (size_t)bp * ldx + coff;
  const float* xj = X + (size_t)(b * P_ + nb) * ldx + coff;
  return (c < Cf) ? xi[c] : (xj[c - Cf] - xi[c - Cf]);
}

// ---------------------------------------------------------------------------
// WMMA GEMM: C[M,N] = A[M,Kd] * W[Kd,N] (+bias).  A: bf16 row-major (EDGE=0)
// or the gathered edge-feature matrix built from fp32 features (EDGE=1).
// Wt: weight pre-transposed to (N x ldt) bf16, zero-padded so Kd-tiles are
// guard-free.  64x64x32 tiles, 128 threads (4 waves), double-buffered LDS.
// M must be a multiple of 64 (E_ and BP_ both are).
// ---------------------------------------------------------------------------
template <int EDGE>
__global__ __launch_bounds__(128)
void gemm_wmma(const __bf16* __restrict__ A, int lda,
               const float* __restrict__ Xsrc, const int* __restrict__ nidx,
               int Cf, int ldx, int coff,
               const __bf16* __restrict__ Wt, int ldw,
               const float* __restrict__ bias,
               float* __restrict__ Cout, int ldc,
               int M, int Kd, int N) {
  __shared__ __bf16 As[2][64][40];
  __shared__ __bf16 Bs[2][64][40];

  const int tid  = threadIdx.x;
  const int lane = tid & 31;
  const int wv   = tid >> 5;
  const int m0   = blockIdx.y * 64;
  const int n0   = blockIdx.x * 64;

  v8f acc[4];
#pragma unroll
  for (int j = 0; j < 4; ++j)
#pragma unroll
    for (int i = 0; i < 8; ++i) acc[j][i] = 0.0f;

  const int mrow = wv * 16 + (lane & 15);
  const int koff = (lane < 16) ? 0 : 8;
  const int T    = (Kd + 31) >> 5;

  // stage t into buffer t&1
  auto stage = [&](int t) {
    int kk = t * 32;
    if (EDGE) {
#pragma unroll
      for (int j = 0; j < 16; ++j) {
        int i = tid + 128 * j;
        int r = i >> 5, c = i & 31;
        int gk = kk + c;
        float v = 0.0f;
        if (gk < Kd) v = edge_elem(m0 + r, gk, Xsrc, nidx, Cf, ldx, coff);
        As[t & 1][r][c] = f2bf(v);
      }
    } else {
      copy_tile(A, lda, m0, kk, As[t & 1], tid);
    }
    copy_tile(Wt, ldw, n0, kk, Bs[t & 1], tid);
  };

  stage(0);
  for (int t = 0; t < T; ++t) {
    if (t + 1 < T) {
      stage(t + 1);
#if USE_ASYNC
      if (EDGE) __builtin_amdgcn_s_wait_asynccnt(2);   // only W async
      else      __builtin_amdgcn_s_wait_asynccnt(4);   // A + W async
#endif
    } else {
#if USE_ASYNC
      __builtin_amdgcn_s_wait_asynccnt(0);
#endif
    }
    __syncthreads();   // stage t visible to all waves

    v16bf af = frag_load(&As[t & 1][mrow][koff]);
#pragma unroll
    for (int j = 0; j < 4; ++j) {
      v16bf bf = frag_load(&Bs[t & 1][j * 16 + (lane & 15)][koff]);
      acc[j] = __builtin_amdgcn_wmma_f32_16x16x32_bf16(
          false, af, false, bf, (short)0, acc[j], false, false);
    }
    __syncthreads();   // all waves done reading buffer t&1
  }

  // D layout: VGPR r -> M = r (+8 for lanes >= 16), N = lane % 16
#pragma unroll
  for (int j = 0; j < 4; ++j) {
    int n = n0 + j * 16 + (lane & 15);
    if (n >= N) continue;
    float bv = bias ? bias[n] : 0.0f;
#pragma unroll
    for (int r = 0; r < 8; ++r) {
      int m = m0 + wv * 16 + r + ((lane < 16) ? 0 : 8);
      if (m < M) Cout[(size_t)m * ldc + n] = acc[j][r] + bv;
    }
  }
}

// ---------------------------------------------------------------------------
// Weight prep: Wt[n*ldt + k] = bf16(W[k*N + n]); zero pad k in [Kd, ldt)
// ---------------------------------------------------------------------------
__global__ __launch_bounds__(256)
void wtrans_bf16(const float* __restrict__ W, __bf16* __restrict__ Wt,
                 int Kd, int N, int ldt) {
  int i = blockIdx.x * 256 + threadIdx.x;
  if (i >= N * ldt) return;
  int n = i / ldt, k = i % ldt;
  float v = (k < Kd) ? W[(size_t)k * N + n] : 0.0f;
  Wt[(size_t)n * ldt + k] = f2bf(v);
}

// ---------------------------------------------------------------------------
// Per-column sum / sumsq (batch-norm stats); sums must be zeroed beforehand.
// ---------------------------------------------------------------------------
__global__ __launch_bounds__(256)
void colstats(const float* __restrict__ X, int R, int C, int ldx,
              float* __restrict__ sums) {
  __shared__ float sh[2][4][64];
  int c  = blockIdx.x * 64 + (threadIdx.x & 63);
  int rg = threadIdx.x >> 6;
  float s = 0.0f, s2 = 0.0f;
  if (c < C) {
    int chunk = (R + gridDim.y - 1) / gridDim.y;
    int r0 = blockIdx.y * chunk;
    int r1 = r0 + chunk; if (r1 > R) r1 = R;
    for (int r = r0 + rg; r < r1; r += 4) {
      float v = X[(size_t)r * ldx + c];
      s += v; s2 += v * v;
    }
  }
  sh[0][rg][threadIdx.x & 63] = s;
  sh[1][rg][threadIdx.x & 63] = s2;
  __syncthreads();
  if (rg == 0 && c < C) {
    int t = threadIdx.x & 63;
    s  = sh[0][0][t] + sh[0][1][t] + sh[0][2][t] + sh[0][3][t];
    s2 = sh[1][0][t] + sh[1][1][t] + sh[1][2][t] + sh[1][3][t];
    atomicAdd(&sums[c], s);
    atomicAdd(&sums[C + c], s2);
  }
}

// lrelu(bn(x)); writes fp32 (Yf) and/or bf16 (Yb), whichever is non-null
__global__ __launch_bounds__(256)
void bn_lrelu_kernel(const float* __restrict__ X, float* __restrict__ Yf,
                     __bf16* __restrict__ Yb, const float* __restrict__ sums,
                     const float* __restrict__ g, const float* __restrict__ bt,
                     int R, int C, int ldx, int ldy, float invR) {
  int c = blockIdx.x * 256 + threadIdx.x;
  if (c >= C) return;
  float m  = sums[c] * invR;
  float v  = fmaxf(sums[C + c] * invR - m * m, 0.0f);
  float sc = rsqrtf(v + EPS_) * g[c];
  float sh = bt[c] - m * sc;
  for (int r = blockIdx.y; r < R; r += gridDim.y) {
    float x = X[(size_t)r * ldx + c] * sc + sh;
    x = (x > 0.0f) ? x : SLOPE_ * x;
    if (Yf) Yf[(size_t)r * ldy + c] = x;
    if (Yb) Yb[(size_t)r * ldy + c] = f2bf(x);
  }
}

// fused normalize + lrelu + max over K; X: (BP*K, C) -> (BP, ldy) at coff
__global__ __launch_bounds__(256)
void bn_lrelu_maxk(const float* __restrict__ X, float* __restrict__ Yf,
                   __bf16* __restrict__ Yb, const float* __restrict__ sums,
                   const float* __restrict__ g, const float* __restrict__ bt,
                   int C, int ldy, int coff, float invR) {
  int c = blockIdx.x * 256 + threadIdx.x;
  if (c >= C) return;
  float m  = sums[c] * invR;
  float v  = fmaxf(sums[C + c] * invR - m * m, 0.0f);
  float sc = rsqrtf(v + EPS_) * g[c];
  float sh = bt[c] - m * sc;
  for (int bp = blockIdx.y; bp < BP_; bp += gridDim.y) {
    float mx = -3.4e38f;
    for (int k = 0; k < K_; ++k) {
      float x = X[((size_t)bp * K_ + k) * C + c] * sc + sh;
      x = (x > 0.0f) ? x : SLOPE_ * x;
      mx = fmaxf(mx, x);
    }
    if (Yf) Yf[(size_t)bp * ldy + coff + c] = mx;
    if (Yb) Yb[(size_t)bp * ldy + coff + c] = f2bf(mx);
  }
}

// max over points: X (B, P, C) -> Y (B, ldy)
__global__ __launch_bounds__(256)
void maxP_kernel(const float* __restrict__ X, float* __restrict__ Y,
                 int C, int ldy) {
  int c = blockIdx.x * 256 + threadIdx.x;
  if (c >= C) return;
  int b = blockIdx.y;
  float mx = -3.4e38f;
  for (int p = 0; p < P_; ++p)
    mx = fmaxf(mx, X[((size_t)b * P_ + p) * C + c]);
  Y[(size_t)b * ldy + c] = mx;
}

// ---------------------------------------------------------------------------
// KNN (top-20 smallest squared distance; ties -> lower index; self included)
// ---------------------------------------------------------------------------
__device__ inline void topk_insert(float d, int c, float* bd, int* bi) {
  if (d >= bd[K_ - 1]) return;
  int j = K_ - 1;
  while (j > 0 && bd[j - 1] > d) { bd[j] = bd[j - 1]; bi[j] = bi[j - 1]; --j; }
  bd[j] = d; bi[j] = c;
}

__global__ __launch_bounds__(256)
void knn3(const float* __restrict__ X, int* __restrict__ out) {
  __shared__ float Pts[P_ * 3];
  int b = blockIdx.x;
  const float* Xb = X + (size_t)b * P_ * 3;
  for (int i = threadIdx.x; i < P_ * 3; i += 256) Pts[i] = Xb[i];
  __syncthreads();
  int p = blockIdx.y * 256 + threadIdx.x;
  float qx = Pts[p * 3 + 0], qy = Pts[p * 3 + 1], qz = Pts[p * 3 + 2];
  float bd[K_]; int bi[K_];
#pragma unroll
  for (int k = 0; k < K_; ++k) { bd[k] = 3.4e38f; bi[k] = 0; }
  for (int c = 0; c < P_; ++c) {
    float dx = qx - Pts[c * 3 + 0];
    float dy = qy - Pts[c * 3 + 1];
    float dz = qz - Pts[c * 3 + 2];
    float d = dx * dx + dy * dy + dz * dz;
    topk_insert(d, c, bd, bi);
  }
  size_t base = ((size_t)b * P_ + p) * K_;
#pragma unroll
  for (int k = 0; k < K_; ++k) out[base + k] = bi[k];
}

__global__ __launch_bounds__(256)
void knn64(const float* __restrict__ X, int ldx, int coff,
           int* __restrict__ out) {
  __shared__ float Cs[128][64];
  int b = blockIdx.x;
  int p = blockIdx.y * 256 + threadIdx.x;
  float q[64];
  const float* qp = X + ((size_t)(b * P_ + p)) * ldx + coff;
#pragma unroll
  for (int i = 0; i < 64; ++i) q[i] = qp[i];
  float bd[K_]; int bi[K_];
#pragma unroll
  for (int k = 0; k < K_; ++k) { bd[k] = 3.4e38f; bi[k] = 0; }
  for (int c0 = 0; c0 < P_; c0 += 128) {
    __syncthreads();
    for (int i = threadIdx.x; i < 128 * 64; i += 256) {
      int r = i >> 6, cc = i & 63;
      Cs[r][cc] = X[((size_t)(b * P_ + c0 + r)) * ldx + coff + cc];
    }
    if (c0 + 128 < P_ && threadIdx.x < 128)
      __builtin_prefetch(X + ((size_t)(b * P_ + c0 + 128 + threadIdx.x)) * ldx + coff, 0, 1);
    __syncthreads();
    for (int r = 0; r < 128; ++r) {
      float d = 0.0f;
#pragma unroll
      for (int i = 0; i < 64; ++i) {
        float df = q[i] - Cs[r][i];
        d = fmaf(df, df, d);
      }
      topk_insert(d, c0 + r, bd, bi);
    }
  }
  size_t base = ((size_t)b * P_ + p) * K_;
#pragma unroll
  for (int k = 0; k < K_; ++k) out[base + k] = bi[k];
}

// ---------------------------------------------------------------------------
// Small kernels (16-row MLPs, T-net tail, transforms, concats)
// ---------------------------------------------------------------------------
__global__ void tiny_mlp(const float* __restrict__ X, const float* __restrict__ W,
                         const float* __restrict__ g, const float* __restrict__ bt,
                         float* __restrict__ Y, int Cin, int Cout) {
  int c = blockIdx.x * blockDim.x + threadIdx.x;
  if (c >= Cout) return;
  float o[16];
  for (int i = 0; i < 16; ++i) {
    float s = 0.0f;
    for (int k = 0; k < Cin; ++k) s = fmaf(X[i * Cin + k], W[(size_t)k * Cout + c], s);
    o[i] = s;
  }
  float m = 0.0f;
#pragma unroll
  for (int i = 0; i < 16; ++i) m += o[i];
  m *= (1.0f / 16.0f);
  float v = 0.0f;
#pragma unroll
  for (int i = 0; i < 16; ++i) { float d = o[i] - m; v += d * d; }
  v *= (1.0f / 16.0f);
  float sc = rsqrtf(v + EPS_) * g[c];
#pragma unroll
  for (int i = 0; i < 16; ++i) {
    float y = (o[i] - m) * sc + bt[c];
    Y[i * Cout + c] = (y > 0.0f) ? y : SLOPE_ * y;
  }
}

__global__ void onehot_mlp(const int* __restrict__ cat, const float* __restrict__ W,
                           const float* __restrict__ g, const float* __restrict__ bt,
                           float* __restrict__ Y) {
  int c = threadIdx.x;   // 64 threads
  float o[16];
#pragma unroll
  for (int i = 0; i < 16; ++i) o[i] = W[(size_t)cat[i] * 64 + c];
  float m = 0.0f;
#pragma unroll
  for (int i = 0; i < 16; ++i) m += o[i];
  m *= (1.0f / 16.0f);
  float v = 0.0f;
#pragma unroll
  for (int i = 0; i < 16; ++i) { float d = o[i] - m; v += d * d; }
  v *= (1.0f / 16.0f);
  float sc = rsqrtf(v + EPS_) * g[c];
#pragma unroll
  for (int i = 0; i < 16; ++i) {
    float y = (o[i] - m) * sc + bt[c];
    Y[i * 64 + c] = (y > 0.0f) ? y : SLOPE_ * y;
  }
}

__global__ void lin9_kernel(const float* __restrict__ G, const float* __restrict__ Wl,
                            const float* __restrict__ bl, float* __restrict__ T) {
  int t = threadIdx.x;
  if (t >= 144) return;
  int i = t / 9, j = t % 9;
  float s = 0.0f;
  for (int k = 0; k < 256; ++k) s = fmaf(G[i * 256 + k], Wl[k * 9 + j], s);
  s += bl[j];
  if (j == 0 || j == 4 || j == 8) s += 1.0f;
  T[i * 9 + j] = s;
}

__global__ void pos_transform(const float* __restrict__ X, const float* __restrict__ T,
                              float* __restrict__ Y) {
  int bp = blockIdx.x * 256 + threadIdx.x;
  if (bp >= BP_) return;
  int b = bp >> 11;
  float x = X[(size_t)bp * 3 + 0], y = X[(size_t)bp * 3 + 1], z = X[(size_t)bp * 3 + 2];
  const float* t = T + b * 9;
#pragma unroll
  for (int d = 0; d < 3; ++d)
    Y[(size_t)bp * 3 + d] = x * t[0 * 3 + d] + y * t[1 * 3 + d] + z * t[2 * 3 + d];
}

__global__ void copy_y(const float* __restrict__ Yy, float* __restrict__ Gc) {
  int t = threadIdx.x;   // 1024 threads
  int row = t >> 6, c = t & 63;
  Gc[row * 1088 + 1024 + c] = Yy[row * 64 + c];
}

// x(BP,1280) bf16 = [broadcast gcat(b,1088), xcat(bp,192)]
__global__ __launch_bounds__(256)
void build_x(const float* __restrict__ Gc, const float* __restrict__ Xc,
             __bf16* __restrict__ Xo) {
  int c = blockIdx.x * 256 + threadIdx.x;
  if (c >= 1280) return;
  for (int bp = blockIdx.y; bp < BP_; bp += gridDim.y) {
    int b = bp >> 11;
    float v = (c < 1088) ? Gc[b * 1088 + c]
                         : Xc[(size_t)bp * 192 + (c - 1088)];
    Xo[(size_t)bp * 1280 + c] = f2bf(v);
  }
}

// ---------------------------------------------------------------------------
// Host orchestration
// ---------------------------------------------------------------------------
extern "C" void kernel_launch(void* const* d_in, const int* in_sizes, int n_in,
                              void* d_out, int out_size, void* d_ws, size_t ws_size,
                              hipStream_t stream) {
  const float* pos = (const float*)d_in[0];
  const int*   cat = (const int*)d_in[1];
  int ii = 2;
  auto F = [&](void) { return (const float*)d_in[ii++]; };
  const float *c1W1=F(), *c1g1=F(), *c1b1=F(), *c1W2=F(), *c1g2=F(), *c1b2=F();
  const float *c2W1=F(), *c2g1=F(), *c2b1=F(), *c2W2=F(), *c2g2=F(), *c2b2=F();
  const float *c3W=F(), *c3g=F(), *c3b=F();
  const float *m1W=F(), *m1g=F(), *m1b=F();
  const float *m2W=F(), *m2g=F(), *m2b=F();
  const float *mfW=F(), *mfb=F();
  const float *h0W=F(), *h0b=F(), *h0g=F(), *h0e=F();
  const float *h1W=F(), *h1b=F(), *h1g=F(), *h1e=F();
  const float *h2W=F(), *h2b=F(), *h2g=F(), *h2e=F();
  const float *tcW1=F(), *tcg1=F(), *tcb1=F(), *tcW2=F(), *tcg2=F(), *tcb2=F();
  const float *tlb=F(), *tlw=F();
  const float *tmW1=F(), *tmg1=F(), *tmb1=F(), *tmW2=F(), *tmg2=F(), *tmb2=F();
  const float *tpW=F(), *tpg=F(), *tpb=F();

  // -------- workspace layout --------
  char* ws = (char*)d_ws;
  size_t off = 0;
  auto take = [&](size_t bytes) -> char* {
    char* p = ws + off;
    off += (bytes + 255) & ~(size_t)255;
    return p;
  };
  int*    idxb  = (int*)   take((size_t)E_ * 4);
  float*  e1    = (float*) take((size_t)E_ * 128 * 4);   // fp32 GEMM outs
  float*  e2    = (float*) take((size_t)E_ * 128 * 4);
  float*  xcat  = (float*) take((size_t)BP_ * 192 * 4);  // fp32 [x1|x2|x3]
  float*  stats = (float*) take(2048 * 4);
  float*  ybuf  = (float*) take(16 * 64 * 4);
  float*  gvec  = (float*) take(16 * 1024 * 4);
  float*  g2    = (float*) take(16 * 512 * 4);
  float*  g3    = (float*) take(16 * 256 * 4);
  float*  gcat  = (float*) take(16 * 1088 * 4);
  float*  tmat  = (float*) take(16 * 9 * 4);
  float*  posT  = (float*) take((size_t)BP_ * 3 * 4);
  float*  m3a   = (float*) take((size_t)BP_ * 256 * 4);
  float*  m3b   = (float*) take((size_t)BP_ * 256 * 4);
  // bf16 activation buffers (A operands of the WMMA GEMMs)
  __bf16* eb    = (__bf16*)take((size_t)E_ * 64 * 2);
  __bf16* xcatb = (__bf16*)take((size_t)BP_ * 192 * 2);
  __bf16* h128b = (__bf16*)take((size_t)BP_ * 128 * 2);
  __bf16* xb    = (__bf16*)take((size_t)BP_ * 1280 * 2);
  __bf16* m3ab  = (__bf16*)take((size_t)BP_ * 256 * 2);
  __bf16* m3bb  = (__bf16*)take((size_t)BP_ * 256 * 2);
  __bf16* m3cb  = (__bf16*)take((size_t)BP_ * 128 * 2);
  (void)ws_size; (void)n_in; (void)in_sizes; (void)out_size;

  // -------- weight prep: bf16, transposed to (N x ldt), zero padded --------
  struct WD { const float* w; int Kd, N; };
  const WD wd[13] = {
      {tcW1, 6, 64},   {tcW2, 64, 128}, {tpW, 128, 1024},
      {c1W1, 6, 64},   {c1W2, 64, 64},  {c2W1, 128, 64},  {c2W2, 64, 64},
      {c3W, 128, 64},  {m1W, 192, 1024},
      {h0W, 1280, 256},{h1W, 256, 256}, {h2W, 256, 128},  {mfW, 128, 50}};
  __bf16* wt[13]; int ldt[13];
  for (int i = 0; i < 13; ++i) {
    ldt[i] = (wd[i].Kd + 31) & ~31;
    size_t n = (size_t)wd[i].N * ldt[i];
    wt[i] = (__bf16*)take(n * 2);
    wtrans_bf16<<<(int)((n + 255) / 256), 256, 0, stream>>>(wd[i].w, wt[i],
                                                            wd[i].Kd, wd[i].N, ldt[i]);
  }
  take(16384);  // guard tail for OOB tile reads on ragged N (N=50 etc.)

  // -------- helpers --------
  auto gemm = [&](const __bf16* A, int lda, const float* Xs, int Cf, int ldx,
                  int coff, int widx, const float* bias, float* Co,
                  int ldc, int M, int Kd, int N, bool edge) {
    dim3 g((N + 63) / 64, (M + 63) / 64), b(128);
    if (edge)
      gemm_wmma<1><<<g, b, 0, stream>>>(A, lda, Xs, idxb, Cf, ldx, coff,
                                        wt[widx], ldt[widx], bias, Co, ldc, M, Kd, N);
    else
      gemm_wmma<0><<<g, b, 0, stream>>>(A, lda, Xs, idxb, Cf, ldx, coff,
                                        wt[widx], ldt[widx], bias, Co, ldc, M, Kd, N);
  };
  auto bnstats = [&](const float* X, int R, int C, int ldx) {
    (void)hipMemsetAsync(stats, 0, 2048 * sizeof(float), stream);
    dim3 g((C + 63) / 64, 128), b(256);
    colstats<<<g, b, 0, stream>>>(X, R, C, ldx, stats);
  };
  auto bn = [&](const float* X, float* Yf, __bf16* Yb, const float* ga,
                const float* be, int R, int C, int ldx, int ldy) {
    dim3 g((C + 255) / 256, 8192), b(256);
    bn_lrelu_kernel<<<g, b, 0, stream>>>(X, Yf, Yb, stats, ga, be, R, C, ldx,
                                         ldy, 1.0f / (float)R);
  };
  auto bnmaxk = [&](const float* X, float* Yf, __bf16* Yb, const float* ga,
                    const float* be, int C, int ldy, int coff) {
    dim3 g((C + 255) / 256, 4096), b(256);
    bn_lrelu_maxk<<<g, b, 0, stream>>>(X, Yf, Yb, stats, ga, be, C, ldy, coff,
                                       1.0f / (float)E_);
  };

  // ===================== pipeline =====================
  onehot_mlp<<<1, 64, 0, stream>>>(cat, m2W, m2g, m2b, ybuf);

  // ---- T-net ----
  knn3<<<dim3(B_, P_ / 256), 256, 0, stream>>>(pos, idxb);
  gemm(nullptr, 0, pos, 3, 3, 0, /*tcW1*/0, nullptr, e1, 64, E_, 6, 64, true);
  bnstats(e1, E_, 64, 64);  bn(e1, nullptr, eb, tcg1, tcb1, E_, 64, 64, 64);
  gemm(eb, 64, nullptr, 0, 0, 0, /*tcW2*/1, nullptr, e2, 128, E_, 64, 128, false);
  bnstats(e2, E_, 128, 128); bnmaxk(e2, nullptr, h128b, tcg2, tcb2, 128, 128, 0);
  gemm(h128b, 128, nullptr, 0, 0, 0, /*tpW*/2, nullptr, e1, 1024, BP_, 128, 1024, false);
  bnstats(e1, BP_, 1024, 1024); bn(e1, e1, nullptr, tpg, tpb, BP_, 1024, 1024, 1024);
  maxP_kernel<<<dim3(4, B_), 256, 0, stream>>>(e1, gvec, 1024, 1024);
  tiny_mlp<<<2, 256, 0, stream>>>(gvec, tmW1, tmg1, tmb1, g2, 1024, 512);
  tiny_mlp<<<1, 256, 0, stream>>>(g2, tmW2, tmg2, tmb2, g3, 512, 256);
  lin9_kernel<<<1, 160, 0, stream>>>(g3, tlw, tlb, tmat);
  pos_transform<<<BP_ / 256, 256, 0, stream>>>(pos, tmat, posT);

  // ---- conv1 ----
  knn3<<<dim3(B_, P_ / 256), 256, 0, stream>>>(posT, idxb);
  gemm(nullptr, 0, posT, 3, 3, 0, /*c1W1*/3, nullptr, e1, 64, E_, 6, 64, true);
  bnstats(e1, E_, 64, 64);  bn(e1, nullptr, eb, c1g1, c1b1, E_, 64, 64, 64);
  gemm(eb, 64, nullptr, 0, 0, 0, /*c1W2*/4, nullptr, e2, 64, E_, 64, 64, false);
  bnstats(e2, E_, 64, 64);  bnmaxk(e2, xcat, xcatb, c1g2, c1b2, 64, 192, 0);   // x1

  // ---- conv2 ----
  knn64<<<dim3(B_, P_ / 256), 256, 0, stream>>>(xcat, 192, 0, idxb);
  gemm(nullptr, 0, xcat, 64, 192, 0, /*c2W1*/5, nullptr, e1, 64, E_, 128, 64, true);
  bnstats(e1, E_, 64, 64);  bn(e1, nullptr, eb, c2g1, c2b1, E_, 64, 64, 64);
  gemm(eb, 64, nullptr, 0, 0, 0, /*c2W2*/6, nullptr, e2, 64, E_, 64, 64, false);
  bnstats(e2, E_, 64, 64);  bnmaxk(e2, xcat, xcatb, c2g2, c2b2, 64, 192, 64);  // x2

  // ---- conv3 ----
  knn64<<<dim3(B_, P_ / 256), 256, 0, stream>>>(xcat, 192, 64, idxb);
  gemm(nullptr, 0, xcat, 64, 192, 64, /*c3W*/7, nullptr, e1, 64, E_, 128, 64, true);
  bnstats(e1, E_, 64, 64);  bnmaxk(e1, xcat, xcatb, c3g, c3b, 64, 192, 128);   // x3

  // ---- mlp1 + global max + concat ----
  gemm(xcatb, 192, nullptr, 0, 0, 0, /*m1W*/8, nullptr, e1, 1024, BP_, 192, 1024, false);
  bnstats(e1, BP_, 1024, 1024); bn(e1, e1, nullptr, m1g, m1b, BP_, 1024, 1024, 1024);
  maxP_kernel<<<dim3(4, B_), 256, 0, stream>>>(e1, gcat, 1024, 1088);
  copy_y<<<1, 1024, 0, stream>>>(ybuf, gcat);
  build_x<<<dim3(5, 8192), 256, 0, stream>>>(gcat, xcat, xb);   // x: (BP,1280) bf16

  // ---- mlp3 head ----
  gemm(xb, 1280, nullptr, 0, 0, 0, /*h0W*/9, h0b, m3a, 256, BP_, 1280, 256, false);
  bnstats(m3a, BP_, 256, 256); bn(m3a, nullptr, m3ab, h0g, h0e, BP_, 256, 256, 256);
  gemm(m3ab, 256, nullptr, 0, 0, 0, /*h1W*/10, h1b, m3b, 256, BP_, 256, 256, false);
  bnstats(m3b, BP_, 256, 256); bn(m3b, nullptr, m3bb, h1g, h1e, BP_, 256, 256, 256);
  gemm(m3bb, 256, nullptr, 0, 0, 0, /*h2W*/11, h2b, m3a, 128, BP_, 256, 128, false);
  bnstats(m3a, BP_, 128, 128); bn(m3a, nullptr, m3cb, h2g, h2e, BP_, 128, 128, 128);
  // final: (BP,128) @ (128,50) + bias -> d_out
  gemm(m3cb, 128, nullptr, 0, 0, 0, /*mfW*/12, mfb, (float*)d_out, 50, BP_, 128, 50, false);
}
